// HeteroRGCN_28020366639260
// MI455X (gfx1250) — matrix operand order repaired and checked by reading, not compiled, not run on hardware
//
#include <hip/hip_runtime.h>

#define D     128
#define NPAT  100000
#define NLAB  2000
#define NDIA  5000
#define NMED  3000
#define E_PL  800000
#define E_PD  400000
#define E_PM  400000
#define LDSP  136   // padded row stride (halves) for LDS weight tile

typedef __attribute__((ext_vector_type(16))) _Float16 v16h;
typedef __attribute__((ext_vector_type(8)))  _Float16 v8h;
typedef __attribute__((ext_vector_type(4)))  _Float16 v4h;
typedef __attribute__((ext_vector_type(8)))  float    v8f;

// ---------------- utility kernels ----------------

__global__ __launch_bounds__(256) void k_zero(float* __restrict__ p, long n) {
  long i = (long)blockIdx.x * blockDim.x + threadIdx.x;
  long st = (long)gridDim.x * blockDim.x;
  for (; i < n; i += st) p[i] = 0.f;
}

// f32 -> f16 cast (vectorized x4; all sizes here are multiples of 128)
__global__ __launch_bounds__(256) void k_cast_h(const float* __restrict__ s,
                                                _Float16* __restrict__ d, long n4) {
  long i = (long)blockIdx.x * blockDim.x + threadIdx.x;
  long st = (long)gridDim.x * blockDim.x;
  for (; i < n4; i += st) {
    float4 v = ((const float4*)s)[i];
    v4h h;
    h[0] = (_Float16)v.x; h[1] = (_Float16)v.y;
    h[2] = (_Float16)v.z; h[3] = (_Float16)v.w;
    ((v4h*)d)[i] = h;
  }
}

// ---- WMMA GEMM: C[M,128] (+)= A[M,128] @ W[128,128]^T (+bias), f16 operands ----
// Block = 8 waves; wave w owns rows [m0+16w, m0+16w+16), all 128 columns.
// W (f16) staged to LDS once per block via async global->LDS loads (ASYNCcnt).
// FLAGS bit0 = add bias, bit1 = accumulate into existing C (compile-time).
template <int FLAGS>
__global__ __launch_bounds__(256) void k_gemm128h(
    const _Float16* __restrict__ A, const _Float16* __restrict__ Wh,
    const float* __restrict__ bias, float* __restrict__ C, int M) {
  __shared__ __align__(16) _Float16 ldsW[D * LDSP];

  // ---- async stage: 128x128 f16 weights -> LDS (8 x b128 per thread) ----
  {
    const int t = threadIdx.x;
    #pragma unroll
    for (int it = 0; it < 8; ++it) {
      int linear = (it << 8) + t;              // 0..2047 -> 2048 x 16B
      int row = linear >> 4, seg = linear & 15;
      unsigned lds_off = (unsigned)(size_t)&ldsW[row * LDSP + (seg << 3)];
      const void* gp = (const void*)(Wh + ((size_t)row << 7) + (seg << 3));
      asm volatile("global_load_async_to_lds_b128 %0, %1, off"
                   :: "v"(lds_off), "v"(gp) : "memory");
    }
    asm volatile("s_wait_asynccnt 0x0" ::: "memory");
  }
  __syncthreads();

  const int lane = threadIdx.x & 31;
  const int wave = threadIdx.x >> 5;
  const int l15  = lane & 15;
  const int half = lane >> 4;
  const int mw = (blockIdx.x << 7) + (wave << 4);   // wave's row tile
  if (mw >= M) return;                               // wave-uniform; M % 16 == 0

  // ---- preload all 4 A chunk fragments (one clause of 8 b128 loads) ----
  // lane holds row mw+l15, K-halves {0..7,16..23} / {8..15,24..31}
  const _Float16* Ap = A + (((size_t)(mw + l15)) << 7) + half * 8;
  v16h av[4];
  #pragma unroll
  for (int kci = 0; kci < 4; ++kci) {
    const v8h alo = *(const v8h*)(Ap + kci * 32);
    const v8h ahi = *(const v8h*)(Ap + kci * 32 + 16);
    av[kci] = __builtin_shufflevector(alo, ahi,
        0, 1, 2, 3, 4, 5, 6, 7, 8, 9, 10, 11, 12, 13, 14, 15);
  }

  // ---- branch-free accumulator init (bias hoisted out of row loop) ----
  v8f c[8];
  #pragma unroll
  for (int jt = 0; jt < 8; ++jt) {
    const int col = (jt << 4) + l15;
    const float bb = (FLAGS & 1) ? bias[col] : 0.f;
    #pragma unroll
    for (int r = 0; r < 8; ++r) {
      float v = bb;
      if (FLAGS & 2) {
        const int row = mw + r + half * 8;
        v += C[((size_t)row << 7) + col];
      }
      c[jt][r] = v;
    }
  }

  // ---- K loop: per chunk, load B-frags in groups of 4, 4 WMMAs per wait ----
  const int brow = l15 * LDSP + (half << 3);
  #pragma unroll
  for (int kci = 0; kci < 4; ++kci) {
    const int kc = kci * 32;
    #pragma unroll
    for (int grp = 0; grp < 2; ++grp) {
      v16h bv[4];
      #pragma unroll
      for (int q = 0; q < 4; ++q) {
        const int jt = grp * 4 + q;
        const int boff = (jt << 4) * LDSP + brow + kc;
        const v8h blo = *(const v8h*)(&ldsW[boff]);
        const v8h bhi = *(const v8h*)(&ldsW[boff + 16]);
        bv[q] = __builtin_shufflevector(blo, bhi,
            0, 1, 2, 3, 4, 5, 6, 7, 8, 9, 10, 11, 12, 13, 14, 15);
      }
      #pragma unroll
      for (int q = 0; q < 4; ++q) {
        const int jt = grp * 4 + q;
        c[jt] = __builtin_amdgcn_wmma_f32_16x16x32_f16(
            false, av[kci], false, bv[q], (short)0, c[jt], false, false);
      }
    }
  }

  #pragma unroll
  for (int jt = 0; jt < 8; ++jt) {
    const int col = (jt << 4) + l15;
    #pragma unroll
    for (int r = 0; r < 8; ++r) {
      const int row = mw + r + half * 8;
      C[((size_t)row << 7) + col] = c[jt][r];
    }
  }
}

// ---------------- edge scatter (segment-sum + counts) ----------------
__global__ __launch_bounds__(256) void k_scatter(
    const float* __restrict__ xs, const int* __restrict__ isrc,
    const int* __restrict__ idst, float* __restrict__ acc,
    float* __restrict__ cnt, int E) {
  int e = (blockIdx.x << 1) + (threadIdx.x >> 7);
  if (e >= E) return;
  int c = threadIdx.x & 127;
  int s = isrc[e], d = idst[e];
  atomicAdd(&acc[(size_t)d * D + c], xs[(size_t)s * D + c]);   // global_atomic_add_f32
  if (c == 0) atomicAdd(&cnt[d], 1.f);
}

__global__ __launch_bounds__(256) void k_divcnt(float* __restrict__ acc,
                                                const float* __restrict__ cnt, long n) {
  long i = (long)blockIdx.x * blockDim.x + threadIdx.x;
  long st = (long)gridDim.x * blockDim.x;
  for (; i < n; i += st) acc[i] /= fmaxf(cnt[i >> 7], 1.f);
}

// ---------------- BatchNorm (batch stats) ----------------
__global__ __launch_bounds__(128) void k_bnstats(const float* __restrict__ Y,
                                                 float* __restrict__ stats, int M) {
  int c = threadIdx.x;
  int r0 = blockIdx.x * 256;
  int r1 = r0 + 256; if (r1 > M) r1 = M;
  float s = 0.f, ss = 0.f;
  for (int r = r0; r < r1; ++r) { float v = Y[(size_t)r * D + c]; s += v; ss += v * v; }
  atomicAdd(&stats[c], s);
  atomicAdd(&stats[D + c], ss);
}

__global__ __launch_bounds__(256) void k_bnrelu(
    const float* __restrict__ Y, float* __restrict__ O,
    const float* __restrict__ stats, const float* __restrict__ gamma,
    const float* __restrict__ beta, int M) {
  long n = (long)M * D;
  long i = (long)blockIdx.x * blockDim.x + threadIdx.x;
  long st = (long)gridDim.x * blockDim.x;
  float invM = 1.f / (float)M;
  for (; i < n; i += st) {
    int c = (int)(i & (D - 1));
    float mean = stats[c] * invM;
    float var = stats[D + c] * invM - mean * mean;
    float v = (Y[i] - mean) * rsqrtf(var + 1e-5f) * gamma[c] + beta[c];
    O[i] = fmaxf(v, 0.f);
  }
}

// ---------------- row L2 normalize ----------------
__global__ __launch_bounds__(128) void k_l2norm(const float* __restrict__ Y,
                                                float* __restrict__ O, int M) {
  __shared__ float sh[128];
  int row = blockIdx.x, t = threadIdx.x;
  float v = Y[(size_t)row * D + t];
  sh[t] = v * v;
  __syncthreads();
  for (int s = 64; s > 0; s >>= 1) {
    if (t < s) sh[t] += sh[t + s];
    __syncthreads();
  }
  float nrm = fmaxf(sqrtf(sh[0]), 1e-12f);
  O[(size_t)row * D + t] = v / nrm;
}

// ---------------- host orchestration ----------------
extern "C" void kernel_launch(void* const* d_in, const int* in_sizes, int n_in,
                              void* d_out, int out_size, void* d_ws, size_t ws_size,
                              hipStream_t stream) {
  (void)in_sizes; (void)n_in; (void)out_size; (void)ws_size;
  const float* emb_p  = (const float*)d_in[0];
  const float* emb_l  = (const float*)d_in[1];
  const float* emb_d  = (const float*)d_in[2];
  const float* emb_m  = (const float*)d_in[3];
  const float* ptW1   = (const float*)d_in[4];
  const float* ptb1   = (const float*)d_in[5];
  const float* ptW2   = (const float*)d_in[6];
  const float* ptb2   = (const float*)d_in[7];
  const float* ptW3   = (const float*)d_in[8];
  const float* ptb3   = (const float*)d_in[9];
  const float* ptg    = (const float*)d_in[10];
  const float* ptbeta = (const float*)d_in[11];
  const float* Wl     = (const float*)d_in[12];
  const float* bl     = (const float*)d_in[13];
  const float* Wr     = (const float*)d_in[14];
  const float* g      = (const float*)d_in[15];
  const float* beta   = (const float*)d_in[16];
  const int* src_pl = (const int*)d_in[17];
  const int* dst_pl = (const int*)d_in[18];
  const int* src_pd = (const int*)d_in[19];
  const int* dst_pd = (const int*)d_in[20];
  const int* src_pm = (const int*)d_in[21];
  const int* dst_pm = (const int*)d_in[22];
  float* outF = (float*)d_out;

  // workspace carve
  char* ws = (char*)d_ws;
  size_t off = 0;
  auto carve = [&](size_t bytes) -> void* {
    void* p = (void*)(ws + off);
    off = (off + bytes + 255) & ~(size_t)255;
    return p;
  };
  const size_t bigB = (size_t)NPAT * D * sizeof(float);
  const size_t bigH = (size_t)NPAT * D * sizeof(_Float16);
  float* bufA  = (float*)carve(bigB);              // mean accumulator / transform ping
  float* bufB  = (float*)carve(bigB);              // new_p accumulator / transform pong
  float* xp    = (float*)carve(bigB);              // patient features (f32)
  float* xlA   = (float*)carve((size_t)NLAB * D * 4);
  float* xlB   = (float*)carve((size_t)NLAB * D * 4);
  float* xdA   = (float*)carve((size_t)NDIA * D * 4);
  float* xdB   = (float*)carve((size_t)NDIA * D * 4);
  float* xmA   = (float*)carve((size_t)NMED * D * 4);
  float* xmB   = (float*)carve((size_t)NMED * D * 4);
  float* cnt   = (float*)carve((size_t)NPAT * 4);
  float* stats = (float*)carve(2 * D * 4);
  _Float16* aH   = (_Float16*)carve(bigH);         // f16 staging for GEMM A operand
  _Float16* cpH  = (_Float16*)carve(bigH);         // f16 x_dst (patients)
  _Float16* clH  = (_Float16*)carve((size_t)NLAB * D * 2);
  _Float16* cdH  = (_Float16*)carve((size_t)NDIA * D * 2);
  _Float16* cmH  = (_Float16*)carve((size_t)NMED * D * 2);
  _Float16* wptH = (_Float16*)carve((size_t)3 * D * D * 2);   // f16 transform weights
  _Float16* wlH  = (_Float16*)carve((size_t)12 * D * D * 2);  // f16 Wl
  _Float16* wrH  = (_Float16*)carve((size_t)12 * D * D * 2);  // f16 Wr

  auto zero = [&](float* p, long n) {
    k_zero<<<dim3((unsigned)((n + 255) / 256)), dim3(256), 0, stream>>>(p, n);
  };
  auto cast = [&](const float* s, _Float16* d, long n) {
    long n4 = n >> 2;
    unsigned gx = (unsigned)((n4 + 255) / 256); if (gx > 4096u) gx = 4096u;
    k_cast_h<<<dim3(gx), dim3(256), 0, stream>>>(s, d, n4);
  };
  auto gemm = [&](const _Float16* A, const _Float16* W, const float* b, float* C, int M, int flags) {
    dim3 gr((unsigned)((M + 127) / 128));
    if (flags == 1)      k_gemm128h<1><<<gr, dim3(256), 0, stream>>>(A, W, b, C, M);
    else if (flags == 2) k_gemm128h<2><<<gr, dim3(256), 0, stream>>>(A, W, b, C, M);
    else                 k_gemm128h<3><<<gr, dim3(256), 0, stream>>>(A, W, b, C, M);
  };
  auto bn = [&](const float* Y, float* O, const float* gm, const float* bt, int M) {
    zero(stats, 2 * D);
    k_bnstats<<<dim3((unsigned)((M + 255) / 256)), dim3(128), 0, stream>>>(Y, stats, M);
    k_bnrelu<<<dim3(1024), dim3(256), 0, stream>>>(Y, O, stats, gm, bt, M);
  };
  auto sage_mean = [&](const float* xs, const int* isrc, const int* idst, int E, int ndst) {
    zero(bufA, (long)ndst * D);
    zero(cnt, ndst);
    k_scatter<<<dim3((unsigned)((E + 1) / 2)), dim3(256), 0, stream>>>(xs, isrc, idst, bufA, cnt, E);
    k_divcnt<<<dim3(2048), dim3(256), 0, stream>>>(bufA, cnt, (long)ndst * D);
    cast(bufA, aH, (long)ndst * D);
  };
  auto WLh = [&](int L, int r) { return wlH + ((size_t)(L * 6 + r)) * D * D; };
  auto WRh = [&](int L, int r) { return wrH + ((size_t)(L * 6 + r)) * D * D; };
  auto BLp = [&](int L, int r) { return bl + (size_t)(L * 6 + r) * D; };
  auto Gp  = [&](int L, int t) { return g    + (size_t)(L * 4 + t) * D; };
  auto Bp  = [&](int L, int t) { return beta + (size_t)(L * 4 + t) * D; };

  // ---- one-time weight casts to f16 ----
  cast(ptW1, wptH, D * D);
  cast(ptW2, wptH + (size_t)D * D, D * D);
  cast(ptW3, wptH + (size_t)2 * D * D, D * D);
  cast(Wl, wlH, (long)12 * D * D);
  cast(Wr, wrH, (long)12 * D * D);

  // ---- patient transform: 2x (GEMM -> BN -> ReLU), GEMM, then L2 normalize ----
  cast(emb_p, aH, (long)NPAT * D);
  gemm(aH, wptH, ptb1, bufA, NPAT, 1);
  bn(bufA, bufA, ptg, ptbeta, NPAT);
  cast(bufA, aH, (long)NPAT * D);
  gemm(aH, wptH + (size_t)D * D, ptb2, bufB, NPAT, 1);
  bn(bufB, bufB, ptg + D, ptbeta + D, NPAT);
  cast(bufB, aH, (long)NPAT * D);
  gemm(aH, wptH + (size_t)2 * D * D, ptb3, bufA, NPAT, 1);
  k_l2norm<<<dim3(NPAT), dim3(128), 0, stream>>>(bufA, xp, NPAT);

  const float *cp = xp, *cl = emb_l, *cd = emb_d, *cm = emb_m;

  for (int L = 0; L < 2; ++L) {
    // f16 copies of current features (x_dst operands, reused across GEMMs)
    cast(cp, cpH, (long)NPAT * D);
    cast(cl, clH, (long)NLAB * D);
    cast(cd, cdH, (long)NDIA * D);
    cast(cm, cmH, (long)NMED * D);

    // new_p = sage(l->p) + sage(d->p) + sage(m->p), accumulated in bufB
    sage_mean(cl, dst_pl, src_pl, E_PL, NPAT);
    gemm(aH,  WLh(L, 1), BLp(L, 1), bufB, NPAT, 1);
    gemm(cpH, WRh(L, 1), nullptr,   bufB, NPAT, 2);
    sage_mean(cd, dst_pd, src_pd, E_PD, NPAT);
    gemm(aH,  WLh(L, 3), BLp(L, 3), bufB, NPAT, 3);
    gemm(cpH, WRh(L, 3), nullptr,   bufB, NPAT, 2);
    sage_mean(cm, dst_pm, src_pm, E_PM, NPAT);
    gemm(aH,  WLh(L, 5), BLp(L, 5), bufB, NPAT, 3);
    gemm(cpH, WRh(L, 5), nullptr,   bufB, NPAT, 2);
    // new_l = sage(p->l)
    sage_mean(cp, src_pl, dst_pl, E_PL, NLAB);
    gemm(aH,  WLh(L, 0), BLp(L, 0), xlA, NLAB, 1);
    gemm(clH, WRh(L, 0), nullptr,   xlA, NLAB, 2);
    // new_d = sage(p->d)
    sage_mean(cp, src_pd, dst_pd, E_PD, NDIA);
    gemm(aH,  WLh(L, 2), BLp(L, 2), xdA, NDIA, 1);
    gemm(cdH, WRh(L, 2), nullptr,   xdA, NDIA, 2);
    // new_m = sage(p->m)
    sage_mean(cp, src_pm, dst_pm, E_PM, NMED);
    gemm(aH,  WLh(L, 4), BLp(L, 4), xmA, NMED, 1);
    gemm(cmH, WRh(L, 4), nullptr,   xmA, NMED, 2);

    const bool last = (L == 1);
    float* oP = last ? outF : xp;
    float* oL = last ? outF + (size_t)NPAT * D : xlB;
    float* oD = last ? outF + (size_t)(NPAT + NLAB) * D : xdB;
    float* oM = last ? outF + (size_t)(NPAT + NLAB + NDIA) * D : xmB;
    bn(bufB, oP, Gp(L, 0), Bp(L, 0), NPAT);
    bn(xlA,  oL, Gp(L, 1), Bp(L, 1), NLAB);
    bn(xdA,  oD, Gp(L, 2), Bp(L, 2), NDIA);
    bn(xmA,  oM, Gp(L, 3), Bp(L, 3), NMED);
    cp = oP; cl = oL; cd = oD; cm = oM;
  }
}
